// Transformer_80315888435346
// MI455X (gfx1250) — matrix operand order repaired and checked by reading, not compiled, not run on hardware
//
#include <hip/hip_runtime.h>
#include <math.h>

// ---------------------------------------------------------------------------
// Model dims (compile-time)
// ---------------------------------------------------------------------------
#define SS    2048      // sequence length
#define DD    1024      // model dim
#define HH    16        // heads
#define HDD   64        // head dim
#define HIDD  2816      // FFN hidden
#define NVOC  32000     // vocab
#define NLAYER 4
#define EPSF  1e-5f

typedef __attribute__((ext_vector_type(16))) __bf16 v16bf;
typedef __attribute__((ext_vector_type(8)))  __bf16 v8bf;
typedef __attribute__((ext_vector_type(4)))  __bf16 v4bf;
typedef __attribute__((ext_vector_type(8)))  float  v8f;

#define FLAG_ACCUM  1   // C f32, accumulate
#define FLAG_TRANS  2   // store transposed
#define FLAG_BF16   4   // store bf16

// ---------------------------------------------------------------------------
// WMMA helper: D = A(16x32 bf16) * B(32x16 bf16) + C(16x16 f32)
// ---------------------------------------------------------------------------
__device__ __forceinline__ v8f wmma_bf16(v16bf a, v16bf b, v8f c) {
  return __builtin_amdgcn_wmma_f32_16x16x32_bf16(false, a, false, b, (short)0, c,
                                                 false, false);
}

// A-matrix 16x32 bf16 fragment from row-major bf16 (row stride ld).
// Lane l: row = l&15 ; K elems = {ko..ko+7, ko+16..ko+23}, ko = (l>=16)*8.
__device__ __forceinline__ v16bf load_a_bf(const __bf16* __restrict__ p, int ld,
                                           int lane) {
  const int row = lane & 15;
  const int ko  = (lane >> 4) << 3;
  const __bf16* r0 = p + (size_t)row * ld + ko;
  v8bf a = *reinterpret_cast<const v8bf*>(r0);
  v8bf b = *reinterpret_cast<const v8bf*>(r0 + 16);
  return __builtin_shufflevector(a, b, 0, 1, 2, 3, 4, 5, 6, 7,
                                 8, 9, 10, 11, 12, 13, 14, 15);
}

// B-matrix 32x16 bf16 fragment. B[k][n] = src[n][k], src row-major stride ld.
// Lane l: col n = l&15 ; K elems = {ko..ko+7, ko+16..ko+23}, ko = (l>=16)*8.
__device__ __forceinline__ v16bf load_b_bf(const __bf16* __restrict__ p, int ld,
                                           int lane) {
  const int row = lane & 15;
  const int ko  = (lane >> 4) << 3;
  const __bf16* r0 = p + (size_t)row * ld + ko;
  v8bf a = *reinterpret_cast<const v8bf*>(r0);
  v8bf b = *reinterpret_cast<const v8bf*>(r0 + 16);
  return __builtin_shufflevector(a, b, 0, 1, 2, 3, 4, 5, 6, 7,
                                 8, 9, 10, 11, 12, 13, 14, 15);
}

// ---------------------------------------------------------------------------
// f32 -> bf16 bulk convert (4 elements / thread)
// ---------------------------------------------------------------------------
__global__ __launch_bounds__(256) void cvt_bf16_kernel(
    const float* __restrict__ in, __bf16* __restrict__ out) {
  int i = (blockIdx.x * 256 + threadIdx.x) * 4;
  float4 v = *reinterpret_cast<const float4*>(in + i);
  v4bf o = {(__bf16)v.x, (__bf16)v.y, (__bf16)v.z, (__bf16)v.w};
  *reinterpret_cast<v4bf*>(out + i) = o;
}

// ---------------------------------------------------------------------------
// Embedding gather: h[s, d] = emb[tokens[s], d]
// ---------------------------------------------------------------------------
__global__ __launch_bounds__(256) void embed_kernel(
    const int* __restrict__ tokens, const float* __restrict__ emb,
    float* __restrict__ h) {
  int idx = blockIdx.x * 256 + threadIdx.x;      // over SS*DD
  int s = idx >> 10;
  int d = idx & (DD - 1);
  h[idx] = emb[(size_t)tokens[s] * DD + d];
}

// ---------------------------------------------------------------------------
// RMSNorm: one block per row; writes bf16 activations
// ---------------------------------------------------------------------------
__global__ __launch_bounds__(256) void rmsnorm_kernel(
    const float* __restrict__ hp, const float* __restrict__ w,
    __bf16* __restrict__ out) {
  __shared__ float red[8];
  const float* row = hp + (size_t)blockIdx.x * DD;
  float ss = 0.f;
#pragma unroll
  for (int j = 0; j < 4; ++j) {
    float v = row[threadIdx.x + j * 256];
    ss += v * v;
  }
#pragma unroll
  for (int m = 16; m >= 1; m >>= 1) ss += __shfl_xor(ss, m, 32);
  if ((threadIdx.x & 31) == 0) red[threadIdx.x >> 5] = ss;
  __syncthreads();
  float tot = 0.f;
#pragma unroll
  for (int i = 0; i < 8; ++i) tot += red[i];
  float inv = rsqrtf(tot * (1.0f / (float)DD) + EPSF);
  __bf16* orow = out + (size_t)blockIdx.x * DD;
#pragma unroll
  for (int j = 0; j < 4; ++j) {
    int i = threadIdx.x + j * 256;
    orow[i] = (__bf16)(row[i] * inv * w[i]);
  }
}

// ---------------------------------------------------------------------------
// Final-row RMSNorm -> f32 (for logits)
// ---------------------------------------------------------------------------
__global__ __launch_bounds__(256) void rmsnorm_f32_kernel(
    const float* __restrict__ hp, const float* __restrict__ w,
    float* __restrict__ out) {
  __shared__ float red[8];
  const float* row = hp;
  float ss = 0.f;
#pragma unroll
  for (int j = 0; j < 4; ++j) {
    float v = row[threadIdx.x + j * 256];
    ss += v * v;
  }
#pragma unroll
  for (int m = 16; m >= 1; m >>= 1) ss += __shfl_xor(ss, m, 32);
  if ((threadIdx.x & 31) == 0) red[threadIdx.x >> 5] = ss;
  __syncthreads();
  float tot = 0.f;
#pragma unroll
  for (int i = 0; i < 8; ++i) tot += red[i];
  float inv = rsqrtf(tot * (1.0f / (float)DD) + EPSF);
#pragma unroll
  for (int j = 0; j < 4; ++j) {
    int i = threadIdx.x + j * 256;
    out[i] = row[i] * inv * w[i];
  }
}

// ---------------------------------------------------------------------------
// GEMM:  C[M,N] (+)= A[M,K] @ W[N,K]^T   (bf16 in, f32 WMMA accumulate)
// Wave owns a 32x64 C tile: 2 A-frags x 4 B-frags -> 8 WMMA per 12 b128 loads.
// Double-buffered k-loop. grid.x = (M/32)*(N/64)/8 , block = 256 (8 waves)
// ---------------------------------------------------------------------------
__global__ __launch_bounds__(256) void gemm_kernel(
    const __bf16* __restrict__ A, int lda,
    const __bf16* __restrict__ W, int ldw,
    void* __restrict__ C, int ldc,
    int M, int N, int K, int flags) {
  const int lane = threadIdx.x & 31;
  const int wave = threadIdx.x >> 5;
  const int t    = blockIdx.x * 8 + wave;
  const int ntn  = N >> 6;
  const int tm   = t / ntn;
  const int tn   = t % ntn;
  if (tm * 32 >= M) return;

  const __bf16* Ab = A + (size_t)tm * 32 * lda;
  const __bf16* Wb = W + (size_t)tn * 64 * ldw;

  v8f acc[8];     // [0..3]: rows 0-15 x 4 col-tiles ; [4..7]: rows 16-31
#pragma unroll
  for (int j = 0; j < 8; ++j)
#pragma unroll
    for (int r = 0; r < 8; ++r) acc[j][r] = 0.f;

  // prologue
  v16bf a0 = load_a_bf(Ab, lda, lane);
  v16bf a1 = load_a_bf(Ab + (size_t)16 * lda, lda, lane);
  v16bf b0 = load_b_bf(Wb, ldw, lane);
  v16bf b1 = load_b_bf(Wb + (size_t)16 * ldw, ldw, lane);
  v16bf b2 = load_b_bf(Wb + (size_t)32 * ldw, ldw, lane);
  v16bf b3 = load_b_bf(Wb + (size_t)48 * ldw, ldw, lane);

  for (int k0 = 32; k0 < K; k0 += 32) {
    // next-iteration fragments issue before current WMMAs consume the old ones
    v16bf an0 = load_a_bf(Ab + k0, lda, lane);
    v16bf an1 = load_a_bf(Ab + (size_t)16 * lda + k0, lda, lane);
    v16bf c0  = load_b_bf(Wb + k0, ldw, lane);
    v16bf c1  = load_b_bf(Wb + (size_t)16 * ldw + k0, ldw, lane);
    v16bf c2  = load_b_bf(Wb + (size_t)32 * ldw + k0, ldw, lane);
    v16bf c3  = load_b_bf(Wb + (size_t)48 * ldw + k0, ldw, lane);
    __builtin_prefetch(Ab + k0 + 32, 0, 1);
    acc[0] = wmma_bf16(a0, b0, acc[0]);
    acc[1] = wmma_bf16(a0, b1, acc[1]);
    acc[2] = wmma_bf16(a0, b2, acc[2]);
    acc[3] = wmma_bf16(a0, b3, acc[3]);
    acc[4] = wmma_bf16(a1, b0, acc[4]);
    acc[5] = wmma_bf16(a1, b1, acc[5]);
    acc[6] = wmma_bf16(a1, b2, acc[6]);
    acc[7] = wmma_bf16(a1, b3, acc[7]);
    a0 = an0; a1 = an1; b0 = c0; b1 = c1; b2 = c2; b3 = c3;
  }
  acc[0] = wmma_bf16(a0, b0, acc[0]);
  acc[1] = wmma_bf16(a0, b1, acc[1]);
  acc[2] = wmma_bf16(a0, b2, acc[2]);
  acc[3] = wmma_bf16(a0, b3, acc[3]);
  acc[4] = wmma_bf16(a1, b0, acc[4]);
  acc[5] = wmma_bf16(a1, b1, acc[5]);
  acc[6] = wmma_bf16(a1, b2, acc[6]);
  acc[7] = wmma_bf16(a1, b3, acc[7]);

  const int rsub = (lane >> 4) << 3;   // row-in-tile = rsub + r (+16 for upper half)
  const int col  = lane & 15;
#pragma unroll
  for (int half = 0; half < 2; ++half) {
#pragma unroll
    for (int j = 0; j < 4; ++j) {
#pragma unroll
      for (int r = 0; r < 8; ++r) {
        int gr = tm * 32 + half * 16 + rsub + r;
        int gc = tn * 64 + j * 16 + col;
        float v = acc[half * 4 + j][r];
        if (flags & FLAG_BF16) {
          __bf16* Cb = (__bf16*)C;
          if (flags & FLAG_TRANS) Cb[(size_t)gc * ldc + gr] = (__bf16)v;
          else                    Cb[(size_t)gr * ldc + gc] = (__bf16)v;
        } else {
          float* Cf = (float*)C;
          float* cp = Cf + (size_t)gr * ldc + gc;
          if (flags & FLAG_ACCUM) *cp += v; else *cp = v;
        }
      }
    }
  }
}

// ---------------------------------------------------------------------------
// RoPE in-place on bf16 [S, H, 64]; pairs (2*d2, 2*d2+1). grid = S*H*32/256
// ---------------------------------------------------------------------------
__global__ __launch_bounds__(256) void rope_kernel(__bf16* __restrict__ x) {
  int idx = blockIdx.x * 256 + threadIdx.x;      // over S*H*32
  int d2 = idx & 31;
  int r  = idx >> 5;
  int hh = r & (HH - 1);
  int s  = r >> 4;
  float inv_freq = expf(-9.210340371976184f * (float)(2 * d2) / (float)HDD);
  float ang = (float)s * inv_freq;
  float c = cosf(ang), sn = sinf(ang);
  __bf16* p = x + (size_t)s * DD + hh * HDD + 2 * d2;
  float xr = (float)p[0], xi = (float)p[1];
  p[0] = (__bf16)(xr * c - xi * sn);
  p[1] = (__bf16)(xr * sn + xi * c);
}

// ---------------------------------------------------------------------------
// Flash attention: one wave per (head, 16-query tile). All operands bf16.
// q,k: [S, D] row-major (post-RoPE). vT: [h*64+d][S]. o: [S, D] bf16.
// grid = H*(S/16)/8 = 256 blocks, block = 256.
// ---------------------------------------------------------------------------
__global__ __launch_bounds__(256) void attn_kernel(
    const __bf16* __restrict__ q, const __bf16* __restrict__ k,
    const __bf16* __restrict__ vT, __bf16* __restrict__ o) {
  __shared__ __bf16 plds[8 * 16 * 32];           // 1KB per wave
  const int lane  = threadIdx.x & 31;
  const int wave  = threadIdx.x >> 5;
  const int t     = blockIdx.x * 8 + wave;
  const int qt    = t & 127;
  const int hh    = t >> 7;
  const int qbase = qt * 16;
  const int rsub  = (lane >> 4) << 3;
  const int col   = lane & 15;

  const __bf16* Qb = q + (size_t)qbase * DD + hh * HDD;
  v16bf qa0 = load_a_bf(Qb, DD, lane);           // head dims 0..31
  v16bf qa1 = load_a_bf(Qb + 32, DD, lane);      // head dims 32..63

  v8f oacc[4];
#pragma unroll
  for (int j = 0; j < 4; ++j)
#pragma unroll
    for (int r = 0; r < 8; ++r) oacc[j][r] = 0.f;
  float mrow[8], lrow[8];
#pragma unroll
  for (int r = 0; r < 8; ++r) { mrow[r] = -1e30f; lrow[r] = 0.f; }

  __bf16* pw = plds + wave * 512;

  for (int kb = 0; kb <= qbase; kb += 32) {
    const __bf16* Kb = k + (size_t)kb * DD + hh * HDD;
    v8f s0, s1;
#pragma unroll
    for (int r = 0; r < 8; ++r) { s0[r] = 0.f; s1[r] = 0.f; }
    // scores for keys kb..kb+15 and kb+16..kb+31 (K = 64 head dims)
    s0 = wmma_bf16(qa0, load_b_bf(Kb, DD, lane), s0);
    s0 = wmma_bf16(qa1, load_b_bf(Kb + 32, DD, lane), s0);
    s1 = wmma_bf16(qa0, load_b_bf(Kb + (size_t)16 * DD, DD, lane), s1);
    s1 = wmma_bf16(qa1, load_b_bf(Kb + (size_t)16 * DD + 32, DD, lane), s1);

    // prefetch V fragments so they are in flight during softmax VALU work
    v16bf vf0 = load_b_bf(vT + (size_t)(hh * HDD +  0) * SS + kb, SS, lane);
    v16bf vf1 = load_b_bf(vT + (size_t)(hh * HDD + 16) * SS + kb, SS, lane);
    v16bf vf2 = load_b_bf(vT + (size_t)(hh * HDD + 32) * SS + kb, SS, lane);
    v16bf vf3 = load_b_bf(vT + (size_t)(hh * HDD + 48) * SS + kb, SS, lane);

    // online softmax (rows live in (reg, lane-half); cols across 16 lanes)
#pragma unroll
    for (int r = 0; r < 8; ++r) {
      int qi  = qbase + rsub + r;
      float v0 = s0[r] * 0.125f;
      float v1 = s1[r] * 0.125f;
      if (kb + col > qi)      v0 = -1e30f;
      if (kb + 16 + col > qi) v1 = -1e30f;
      float mx = fmaxf(v0, v1);
#pragma unroll
      for (int m = 8; m >= 1; m >>= 1) mx = fmaxf(mx, __shfl_xor(mx, m, 32));
      float mnew = fmaxf(mrow[r], mx);
      float scl  = __expf(mrow[r] - mnew);
      float p0   = __expf(v0 - mnew);
      float p1   = __expf(v1 - mnew);
      float ps   = p0 + p1;
#pragma unroll
      for (int m = 8; m >= 1; m >>= 1) ps += __shfl_xor(ps, m, 32);
      lrow[r] = lrow[r] * scl + ps;
      mrow[r] = mnew;
#pragma unroll
      for (int j = 0; j < 4; ++j) oacc[j][r] *= scl;
      int prow = rsub + r;
      pw[prow * 32 + col]      = (__bf16)p0;
      pw[prow * 32 + col + 16] = (__bf16)p1;
    }

    // reload P as A-fragment (16x32) from LDS
    v16bf pf;
    {
      int prow = lane & 15;
      int ko   = (lane >> 4) << 3;
      v8bf pa = *reinterpret_cast<const v8bf*>(pw + prow * 32 + ko);
      v8bf pb = *reinterpret_cast<const v8bf*>(pw + prow * 32 + ko + 16);
      pf = __builtin_shufflevector(pa, pb, 0, 1, 2, 3, 4, 5, 6, 7,
                                   8, 9, 10, 11, 12, 13, 14, 15);
    }
    oacc[0] = wmma_bf16(pf, vf0, oacc[0]);
    oacc[1] = wmma_bf16(pf, vf1, oacc[1]);
    oacc[2] = wmma_bf16(pf, vf2, oacc[2]);
    oacc[3] = wmma_bf16(pf, vf3, oacc[3]);
  }

#pragma unroll
  for (int j = 0; j < 4; ++j)
#pragma unroll
    for (int r = 0; r < 8; ++r) {
      int gr = qbase + rsub + r;
      int gc = hh * HDD + j * 16 + col;
      o[(size_t)gr * DD + gc] = (__bf16)(oacc[j][r] / lrow[r]);
    }
}

// ---------------------------------------------------------------------------
// g1b = bf16( silu(g1) * g3 )   (elementwise over S*HID)
// ---------------------------------------------------------------------------
__global__ __launch_bounds__(256) void silu_mul_kernel(
    const float* __restrict__ g1, const float* __restrict__ g3,
    __bf16* __restrict__ g1b) {
  int i = blockIdx.x * 256 + threadIdx.x;
  float a = g1[i];
  g1b[i] = (__bf16)((a / (1.f + __expf(-a))) * g3[i]);
}

// ---------------------------------------------------------------------------
// logits[v] = dot(xn, emb[v])  -- one wave per vocab row
// ---------------------------------------------------------------------------
__global__ __launch_bounds__(256) void logits_kernel(
    const float* __restrict__ xn, const float* __restrict__ emb,
    float* __restrict__ out) {
  const int lane = threadIdx.x & 31;
  const int vid  = (blockIdx.x * 256 + threadIdx.x) >> 5;
  if (vid >= NVOC) return;
  const float* e = emb + (size_t)vid * DD;
  float acc = 0.f;
#pragma unroll 4
  for (int i = lane; i < DD; i += 32) acc += e[i] * xn[i];
#pragma unroll
  for (int m = 16; m >= 1; m >>= 1) acc += __shfl_xor(acc, m, 32);
  if (lane == 0) out[vid] = acc;
}

// ---------------------------------------------------------------------------
// Host orchestration
// ---------------------------------------------------------------------------
extern "C" void kernel_launch(void* const* d_in, const int* in_sizes, int n_in,
                              void* d_out, int out_size, void* d_ws, size_t ws_size,
                              hipStream_t stream) {
  (void)in_sizes; (void)n_in; (void)out_size; (void)ws_size;
  const int*   tokens = (const int*)d_in[0];
  const float* emb    = (const float*)d_in[1];
  const float* wq     = (const float*)d_in[2];
  const float* wk     = (const float*)d_in[3];
  const float* wv     = (const float*)d_in[4];
  const float* wo     = (const float*)d_in[5];
  const float* w1     = (const float*)d_in[6];
  const float* w2     = (const float*)d_in[7];
  const float* w3     = (const float*)d_in[8];
  const float* anw    = (const float*)d_in[9];
  const float* fnw    = (const float*)d_in[10];
  const float* finw   = (const float*)d_in[11];
  float* out = (float*)d_out;

  // --- workspace layout ----------------------------------------------------
  const size_t SD  = (size_t)SS * DD;      // 2,097,152
  const size_t SH  = (size_t)SS * HIDD;    // 5,767,168
  const size_t DxD = (size_t)DD * DD;      // 1,048,576
  const size_t HxD = (size_t)HIDD * DD;    // 2,883,584

  float* h  = (float*)d_ws;                // f32 residual
  float* g1 = h  + SD;                     // f32 FFN gate
  float* g3 = g1 + SH;                     // f32 FFN up
  float* xn = g3 + SH;                     // f32 final-row norm (D)
  __bf16* xbf = (__bf16*)(xn + DD);        // bf16 normed activations
  __bf16* qb  = xbf + SD;
  __bf16* kb  = qb  + SD;
  __bf16* vTb = kb  + SD;
  __bf16* ob  = vTb + SD;
  __bf16* g1b = ob  + SD;                  // bf16 silu(g1)*g3
  __bf16* wqb = g1b + SH;                  // per-layer bf16 weights
  __bf16* wkb = wqb + DxD;
  __bf16* wvb = wkb + DxD;
  __bf16* wob = wvb + DxD;
  __bf16* w1b = wob + DxD;
  __bf16* w2b = w1b + HxD;
  __bf16* w3b = w2b + HxD;

  const dim3 blk(256);
  const int gemm_d   = (SS / 32) * (DD / 64) / 8;     // 128
  const int gemm_hid = (SS / 32) * (HIDD / 64) / 8;   // 352
  const int cvt_dd   = (int)(DxD / 1024);             // 1024
  const int cvt_hd   = (int)(HxD / 1024);             // 2816

  embed_kernel<<<SS * DD / 256, blk, 0, stream>>>(tokens, emb, h);

  for (int l = 0; l < NLAYER; ++l) {
    // stage this layer's weights as bf16
    cvt_bf16_kernel<<<cvt_dd, blk, 0, stream>>>(wq + l * DxD, wqb);
    cvt_bf16_kernel<<<cvt_dd, blk, 0, stream>>>(wk + l * DxD, wkb);
    cvt_bf16_kernel<<<cvt_dd, blk, 0, stream>>>(wv + l * DxD, wvb);
    cvt_bf16_kernel<<<cvt_dd, blk, 0, stream>>>(wo + l * DxD, wob);
    cvt_bf16_kernel<<<cvt_hd, blk, 0, stream>>>(w1 + l * HxD, w1b);
    cvt_bf16_kernel<<<cvt_hd, blk, 0, stream>>>(w2 + l * HxD, w2b);
    cvt_bf16_kernel<<<cvt_hd, blk, 0, stream>>>(w3 + l * HxD, w3b);

    rmsnorm_kernel<<<SS, blk, 0, stream>>>(h, anw + (size_t)l * DD, xbf);
    gemm_kernel<<<gemm_d, blk, 0, stream>>>(xbf, DD, wqb, DD, qb,  DD, SS, DD, DD, FLAG_BF16);
    gemm_kernel<<<gemm_d, blk, 0, stream>>>(xbf, DD, wkb, DD, kb,  DD, SS, DD, DD, FLAG_BF16);
    gemm_kernel<<<gemm_d, blk, 0, stream>>>(xbf, DD, wvb, DD, vTb, SS, SS, DD, DD,
                                            FLAG_BF16 | FLAG_TRANS);
    rope_kernel<<<SS * HH * 32 / 256, blk, 0, stream>>>(qb);
    rope_kernel<<<SS * HH * 32 / 256, blk, 0, stream>>>(kb);
    attn_kernel<<<HH * (SS / 16) / 8, blk, 0, stream>>>(qb, kb, vTb, ob);
    gemm_kernel<<<gemm_d, blk, 0, stream>>>(ob, DD, wob, DD, h, DD, SS, DD, DD, FLAG_ACCUM);

    rmsnorm_kernel<<<SS, blk, 0, stream>>>(h, fnw + (size_t)l * DD, xbf);
    gemm_kernel<<<gemm_hid, blk, 0, stream>>>(xbf, DD, w1b, DD, g1, HIDD, SS, HIDD, DD, 0);
    gemm_kernel<<<gemm_hid, blk, 0, stream>>>(xbf, DD, w3b, DD, g3, HIDD, SS, HIDD, DD, 0);
    silu_mul_kernel<<<SS * HIDD / 256, blk, 0, stream>>>(g1, g3, g1b);
    gemm_kernel<<<gemm_d, blk, 0, stream>>>(g1b, HIDD, w2b, HIDD, h, DD, SS, DD, HIDD,
                                            FLAG_ACCUM);
  }

  rmsnorm_f32_kernel<<<1, blk, 0, stream>>>(h + (size_t)(SS - 1) * DD, finw, xn);
  logits_kernel<<<NVOC / 8, blk, 0, stream>>>(xn, emb, out);
}
